// FP8PureLinear_47321949667427
// MI455X (gfx1250) — 8-bit matrix operand order repaired and checked by reading, compiled, not run on hardware
//
#include <hip/hip_runtime.h>


#define M_TOT   65536
#define K_TOT   512
#define N_TOT   512

typedef int   v16i __attribute__((ext_vector_type(16)));
typedef float v8f  __attribute__((ext_vector_type(8)));

__device__ __forceinline__ float bflo(unsigned u) {
    return __builtin_bit_cast(float, u << 16);
}
__device__ __forceinline__ float bfhi(unsigned u) {
    return __builtin_bit_cast(float, u & 0xffff0000u);
}
__device__ __forceinline__ unsigned short f2bf16(float f) {
    unsigned u = __builtin_bit_cast(unsigned, f);
    unsigned r = u + 0x7fffu + ((u >> 16) & 1u);
    return (unsigned short)(r >> 16);
}

__device__ __forceinline__ unsigned pack2_bf16(float a, float b) {
#if defined(__gfx1250__) && __has_builtin(__builtin_amdgcn_cvt_pk_bf16_f32)
    typedef __bf16 v2bf __attribute__((ext_vector_type(2)));
    v2bf r = __builtin_amdgcn_cvt_pk_bf16_f32(a, b);
    return __builtin_bit_cast(unsigned, r);
#else
    return (unsigned)f2bf16(a) | ((unsigned)f2bf16(b) << 16);
#endif
}

__device__ __forceinline__ unsigned fp8e4m3_sw(float f) {
    unsigned s = (__builtin_bit_cast(unsigned, f) >> 31) << 7;
    float a = fabsf(f);
    if (!(a > 0.0f)) return s;
    if (a > 448.0f) a = 448.0f;
    int e; float m = frexpf(a, &e);
    int E = e - 1;
    if (E < -6) {
        unsigned q = (unsigned)rintf(a * 512.0f);
        if (q > 7u) q = 7u;
        return s | q;
    }
    unsigned mant = (unsigned)rintf((m * 2.0f - 1.0f) * 8.0f);
    if (mant == 8u) { mant = 0u; ++E; }
    if (E > 8) { E = 8; mant = 7u; }
    return s | ((unsigned)(E + 7) << 3) | mant;
}

__device__ __forceinline__ int pack4_fp8(float a, float b, float c, float d) {
#if defined(__gfx1250__) && __has_builtin(__builtin_amdgcn_cvt_pk_fp8_f32)
    int v = 0;
    v = __builtin_amdgcn_cvt_pk_fp8_f32(a, b, v, false);
    v = __builtin_amdgcn_cvt_pk_fp8_f32(c, d, v, true);
    return v;
#else
    return (int)(fp8e4m3_sw(a) | (fp8e4m3_sw(b) << 8) |
                 (fp8e4m3_sw(c) << 16) | (fp8e4m3_sw(d) << 24));
#endif
}

__device__ __forceinline__ float e4m3_scale(float amax) {
    return fmaxf(amax * (1.0f / 448.0f), 1e-12f);
}

__global__ void fp8lin_init_kernel(unsigned* __restrict__ amax) {
    if (threadIdx.x < 2) amax[threadIdx.x] = 0u;
}

__global__ __launch_bounds__(256)
void fp8lin_amax_kernel(const uint4* __restrict__ p, long long n4,
                        unsigned* __restrict__ amax_out) {
    float m = 0.0f;
    long long i      = (long long)blockIdx.x * blockDim.x + threadIdx.x;
    long long stride = (long long)gridDim.x * blockDim.x;
    for (; i < n4; i += stride) {
        const uint4 q = p[i];
        m = fmaxf(m, fmaxf(fabsf(bflo(q.x)), fabsf(bfhi(q.x))));
        m = fmaxf(m, fmaxf(fabsf(bflo(q.y)), fabsf(bfhi(q.y))));
        m = fmaxf(m, fmaxf(fabsf(bflo(q.z)), fabsf(bfhi(q.z))));
        m = fmaxf(m, fmaxf(fabsf(bflo(q.w)), fabsf(bfhi(q.w))));
    }
    __shared__ float red[256];
    red[threadIdx.x] = m;
    __syncthreads();
    for (int s = 128; s > 0; s >>= 1) {
        if ((int)threadIdx.x < s)
            red[threadIdx.x] = fmaxf(red[threadIdx.x], red[threadIdx.x + s]);
        __syncthreads();
    }
    if (threadIdx.x == 0)
        atomicMax(amax_out, __builtin_bit_cast(unsigned, red[0]));
}

__global__ __launch_bounds__(256)
void fp8lin_quantw_kernel(const unsigned* __restrict__ w2,
                          unsigned* __restrict__ w8,
                          const unsigned* __restrict__ amax_w) {
    float r = 1.0f / e4m3_scale(__builtin_bit_cast(float, *amax_w));
    int i = blockIdx.x * blockDim.x + threadIdx.x;
    unsigned u0 = w2[2 * i];
    unsigned u1 = w2[2 * i + 1];
    w8[i] = (unsigned)pack4_fp8(bflo(u0) * r, bfhi(u0) * r,
                                bflo(u1) * r, bfhi(u1) * r);
}

__global__ __launch_bounds__(256)
void fp8lin_gemm_kernel(const unsigned* __restrict__ x2,
                        const unsigned char* __restrict__ w8,
                        const unsigned* __restrict__ amax,
                        unsigned short* __restrict__ out) {
    const int lane = threadIdx.x & 31;
    const int wid  = threadIdx.x >> 5;
    const int wm   = wid & 1;
    const int wn   = wid >> 1;
    const int lo16 = lane & 15;
    const int hi   = lane >> 4;

    const float sx = e4m3_scale(__builtin_bit_cast(float, amax[0]));
    const float sw = e4m3_scale(__builtin_bit_cast(float, amax[1]));
    const float rx = 1.0f / sx;
    const float oscale = sx * sw;

    const int rowA     = blockIdx.x * 32 + wm * 16 + lo16;
    const int ncolbase = wn * 128;

    const unsigned* xrow = x2 + (size_t)rowA * (K_TOT / 2);

    v8f acc[8];
#pragma unroll
    for (int t = 0; t < 8; ++t) acc[t] = (v8f)0.0f;

#pragma unroll
    for (int ks = 0; ks < 4; ++ks) {
        const int K0 = ks * 128;
        const int kb = K0 + hi * 16;

        v16i A;
#pragma unroll
        for (int j = 0; j < 8; ++j) {
            const uint4 q = *(const uint4*)(xrow + ((kb + (j >> 1) * 32 + (j & 1) * 8) >> 1));
            A[2 * j]     = pack4_fp8(bflo(q.x) * rx, bfhi(q.x) * rx,
                                     bflo(q.y) * rx, bfhi(q.y) * rx);
            A[2 * j + 1] = pack4_fp8(bflo(q.z) * rx, bfhi(q.z) * rx,
                                     bflo(q.w) * rx, bfhi(q.w) * rx);
        }

#pragma unroll
        for (int t = 0; t < 8; ++t) {
            const int ncol = ncolbase + t * 16 + lo16;
            const unsigned char* wp = w8 + (size_t)ncol * K_TOT + K0 + hi * 16;
            v16i B;
#pragma unroll
            for (int j = 0; j < 4; ++j) {
                const uint4 q = *(const uint4*)(wp + j * 32);
                B[4 * j]     = (int)q.x;
                B[4 * j + 1] = (int)q.y;
                B[4 * j + 2] = (int)q.z;
                B[4 * j + 3] = (int)q.w;
            }
            acc[t] = __builtin_amdgcn_wmma_f32_16x16x128_fp8_fp8(
                         A, B, (short)0, acc[t],
                          false,  false);
        }
    }

    const int rbase = blockIdx.x * 32 + wm * 16 + hi * 8;
#pragma unroll
    for (int t = 0; t < 8; ++t) {
        const int ncol = ncolbase + t * 16 + lo16;
#pragma unroll
        for (int r = 0; r < 8; r += 2) {
            const unsigned p = pack2_bf16(acc[t][r] * oscale,
                                          acc[t][r + 1] * oscale);
            out[(size_t)(rbase + r)     * N_TOT + ncol] = (unsigned short)(p & 0xffffu);
            out[(size_t)(rbase + r + 1) * N_TOT + ncol] = (unsigned short)(p >> 16);
        }
    }
}

extern "C" void kernel_launch(void* const* d_in, const int* in_sizes, int n_in,
                              void* d_out, int out_size, void* d_ws, size_t ws_size,
                              hipStream_t stream) {
    (void)in_sizes; (void)n_in; (void)out_size; (void)ws_size;

    const unsigned* x2 = (const unsigned*)d_in[0];
    const unsigned* w2 = (const unsigned*)d_in[1];
    unsigned short* out = (unsigned short*)d_out;

    unsigned*      amax = (unsigned*)d_ws;
    unsigned char* w8   = (unsigned char*)d_ws + 256;

    const long long x_quads = (long long)M_TOT * (K_TOT / 8);
    const long long w_quads = (long long)N_TOT * (K_TOT / 8);

    fp8lin_init_kernel<<<1, 64, 0, stream>>>(amax);
    fp8lin_amax_kernel<<<2048, 256, 0, stream>>>((const uint4*)x2, x_quads, amax + 0);
    fp8lin_amax_kernel<<<32,   256, 0, stream>>>((const uint4*)w2, w_quads, amax + 1);
    fp8lin_quantw_kernel<<<(N_TOT * K_TOT / 4) / 256, 256, 0, stream>>>(w2, (unsigned*)w8, amax + 1);
    fp8lin_gemm_kernel<<<M_TOT / 32, 256, 0, stream>>>(x2, w8, amax, out);
}
